// BILSTM_INIT_STATE_MH_ATTN_84335977824394
// MI455X (gfx1250) — matrix operand order repaired and checked by reading, not compiled, not run on hardware
//
#include <hip/hip_runtime.h>
#include <hip/hip_bf16.h>

typedef __attribute__((ext_vector_type(16))) _Float16 v16h;
typedef __attribute__((ext_vector_type(8)))  _Float16 v8h;
typedef __attribute__((ext_vector_type(8)))  float    v8f;
typedef __attribute__((ext_vector_type(2)))  float    f2;

#define B_  256
#define T_  512

__device__ __forceinline__ float sigf(float x) { return 1.0f / (1.0f + __expf(-x)); }

__device__ __forceinline__ v8f wmma16(v16h a, v16h b, v8f c) {
  return __builtin_amdgcn_wmma_f32_16x16x32_f16(false, a, false, b, (short)0, c, false, false);
}

// A-fragment: 16x32 f16 tile, row-major source, row = lane&15.
// Per ISA: lane<16 holds K={0..7,16..23}; lane>=16 holds K={8..15,24..31}.
__device__ __forceinline__ v16h fragA_f16(const _Float16* base, long stride, int lane) {
  int r = lane & 15, hi = lane >> 4;
  const _Float16* p = base + (long)r * stride + hi * 8;
  union { v16h v; v8h h[2]; } u;
  u.h[0] = *(const v8h*)(p);
  u.h[1] = *(const v8h*)(p + 16);
  return u.v;
}

// B-fragment: 32x16 (KxN) built from weight W stored row-major [N][K]:
// lane holds col n = lane&15; lane<16 -> K={0..7,16..23}, lane>=16 -> K={8..15,24..31}.
__device__ __forceinline__ v16h fragB_f16(const _Float16* base, long stride, int lane) {
  int n = lane & 15, hi = lane >> 4;
  const _Float16* p = base + (long)n * stride + hi * 8;
  union { v16h v; v8h h[2]; } u;
  u.h[0] = *(const v8h*)(p);
  u.h[1] = *(const v8h*)(p + 16);
  return u.v;
}

__device__ __forceinline__ v16h fragB_f32cvt(const float* base, int stride, int lane) {
  int n = lane & 15, hi = lane >> 4;
  const float* p = base + n * stride + hi * 8;
  v16h v;
#pragma unroll
  for (int j = 0; j < 16; ++j) v[j] = (_Float16)p[j + (j & 8)];
  return v;
}

// ---------------------------------------------------------------------------
// K1: h0/c0 init: [4][256][32], k order = [first,last,first,last]
// ---------------------------------------------------------------------------
__global__ __launch_bounds__(256) void k_init(
    const float* __restrict__ xcf, const float* __restrict__ xcl,
    const float* __restrict__ h0w, const float* __restrict__ h0b,
    const float* __restrict__ c0w, const float* __restrict__ c0b,
    float* __restrict__ h0o, float* __restrict__ c0o) {
  int idx = blockIdx.x * 256 + threadIdx.x;        // 0..32767
  int n = idx & 31;
  int b = (idx >> 5) & 255;
  int k = idx >> 13;
  const float* xc = (k & 1) ? (xcl + b * 49) : (xcf + b * 49);
  const float* wh = h0w + (k * 32 + n) * 49;
  const float* wc = c0w + (k * 32 + n) * 49;
  float sh = h0b[k * 32 + n], sc = c0b[k * 32 + n];
#pragma unroll
  for (int d = 0; d < 49; ++d) { sh += wh[d] * xc[d]; sc += wc[d] * xc[d]; }
  h0o[idx] = sh;
  c0o[idx] = sc;
}

// ---------------------------------------------------------------------------
// K2: h_tag[B*T,32] = f16(x_tag @ pre_w^T + pre_b)   M=131072, N=32, K=194->224
// 8 waves/block, each wave one 16-row M-tile, two 16x16 N-tiles.
// Chunks k0=0..160 are fully in-range (max k = 191 < 194): unguarded float2
// vector loads (row*194 is even -> 8B aligned). Tail chunk k0=192: only
// lanes hi==0 hold valid elements (k=192,193 at j=0,1).
// ---------------------------------------------------------------------------
__global__ __launch_bounds__(256) void k_htag(
    const float* __restrict__ x_tag, const float* __restrict__ pre_w,
    const float* __restrict__ pre_b, _Float16* __restrict__ htag) {
  __shared__ _Float16 bw[32 * 224];
  int tid = threadIdx.x;
  for (int i = tid; i < 32 * 224; i += 256) {
    int n = i / 224, k = i % 224;
    bw[i] = (k < 194) ? (_Float16)pre_w[n * 194 + k] : (_Float16)0.0f;
  }
  __syncthreads();
  int wid = tid >> 5, lane = tid & 31;
  long row0 = ((long)blockIdx.x * 8 + wid) * 16;     // 8192 tiles
  int r = lane & 15, hi = lane >> 4;
  const float* ap = x_tag + (row0 + r) * 194;
  v8f acc0 = {}; v8f acc1 = {};
#pragma unroll
  for (int k0 = 0; k0 < 192; k0 += 32) {
    v16h A;
    const f2* pa = (const f2*)(ap + k0 + hi * 8);
    const f2* pb = (const f2*)(ap + k0 + 16 + hi * 8);
#pragma unroll
    for (int j = 0; j < 4; ++j) {
      f2 a = pa[j], b = pb[j];
      A[2 * j]     = (_Float16)a[0];
      A[2 * j + 1] = (_Float16)a[1];
      A[8 + 2 * j]     = (_Float16)b[0];
      A[8 + 2 * j + 1] = (_Float16)b[1];
    }
    acc0 = wmma16(A, fragB_f16(bw + k0, 224, lane), acc0);
    acc1 = wmma16(A, fragB_f16(bw + 16 * 224 + k0, 224, lane), acc1);
  }
  {
    v16h A = {};
    if (hi == 0) {
      f2 a = *(const f2*)(ap + 192);
      A[0] = (_Float16)a[0];
      A[1] = (_Float16)a[1];
    }
    acc0 = wmma16(A, fragB_f16(bw + 192, 224, lane), acc0);
    acc1 = wmma16(A, fragB_f16(bw + 16 * 224 + 192, 224, lane), acc1);
  }
  float pb0 = pre_b[r], pb1 = pre_b[16 + r];
#pragma unroll
  for (int rr = 0; rr < 8; ++rr) {
    long row = row0 + rr + 8 * hi;
    htag[row * 32 + r]      = (_Float16)(acc0[rr] + pb0);
    htag[row * 32 + 16 + r] = (_Float16)(acc1[rr] + pb1);
  }
}

// ---------------------------------------------------------------------------
// K3: LSTM layer 0.  grid(16 batch-tiles, 2 dirs), 128 threads = 4 waves,
// wave w owns gate w (cols 32w..32w+31 = two 16x16 N-tiles).
// ---------------------------------------------------------------------------
__global__ __launch_bounds__(128) void k_lstm0(
    const _Float16* __restrict__ htag,      // [B][T][32] f16
    const float* __restrict__ rnn0_wih,     // [2][128][32]
    const float* __restrict__ rnn_whh,      // [2][2][128][32]
    const float* __restrict__ rnn_bih,      // [2][2][128]
    const float* __restrict__ rnn_bhh,
    const float* __restrict__ h0, const float* __restrict__ c0,  // [4][256][32]
    _Float16* __restrict__ x1)              // [B][T][64] f16
{
  const int dir = blockIdx.y;
  const int b0 = blockIdx.x * 16;
  const int tid = threadIdx.x, lane = tid & 31, wid = tid >> 5;
  const float* wih = rnn0_wih + dir * 128 * 32;
  const float* whh = rnn_whh + dir * 128 * 32;      // layer 0
  const float* bih = rnn_bih + dir * 128;
  const float* bhh = rnn_bhh + dir * 128;

  __shared__ _Float16 hbuf[16 * 32];
  __shared__ float    gbuf[16 * 128];

  const int n0 = wid * 32;
  const int r = lane & 15, hi = lane >> 4;
  v16h Bx0 = fragB_f32cvt(wih + n0 * 32, 32, lane);
  v16h Bx1 = fragB_f32cvt(wih + (n0 + 16) * 32, 32, lane);
  v16h Bh0 = fragB_f32cvt(whh + n0 * 32, 32, lane);
  v16h Bh1 = fragB_f32cvt(whh + (n0 + 16) * 32, 32, lane);
  float bias0 = bih[n0 + r] + bhh[n0 + r];
  float bias1 = bih[n0 + 16 + r] + bhh[n0 + 16 + r];

  // state: each thread owns 4 (m,n) elements
  const int e0 = tid * 4, em = e0 >> 5, en = e0 & 31;
  float c[4];
  const float* c0p = c0 + (dir * B_ + b0) * 32;     // k = dir (layer 0)
  const float* h0p = h0 + (dir * B_ + b0) * 32;
#pragma unroll
  for (int j = 0; j < 4; ++j) {
    c[j] = c0p[em * 32 + en + j];
    hbuf[em * 32 + en + j] = (_Float16)h0p[em * 32 + en + j];
  }

  for (int s = 0; s < T_; ++s) {
    const int t = dir ? (T_ - 1 - s) : s;
    __syncthreads();
    v16h Ax = fragA_f16(htag + ((long)b0 * T_ + t) * 32, (long)T_ * 32, lane);
    v16h Ah = fragA_f16(hbuf, 32, lane);
    v8f a0 = {}; v8f a1 = {};
    a0 = wmma16(Ax, Bx0, a0);
    a1 = wmma16(Ax, Bx1, a1);
    a0 = wmma16(Ah, Bh0, a0);
    a1 = wmma16(Ah, Bh1, a1);
#pragma unroll
    for (int rr = 0; rr < 8; ++rr) {
      gbuf[(rr + 8 * hi) * 128 + n0 + r]      = a0[rr] + bias0;
      gbuf[(rr + 8 * hi) * 128 + n0 + 16 + r] = a1[rr] + bias1;
    }
    __syncthreads();
#pragma unroll
    for (int j = 0; j < 4; ++j) {
      int n = en + j;
      float gi = gbuf[em * 128 + n];
      float gf = gbuf[em * 128 + 32 + n];
      float gg = gbuf[em * 128 + 64 + n];
      float go = gbuf[em * 128 + 96 + n];
      float cc = sigf(gf) * c[j] + sigf(gi) * tanhf(gg);
      float hh = sigf(go) * tanhf(cc);
      c[j] = cc;
      hbuf[em * 32 + n] = (_Float16)hh;
      x1[((long)(b0 + em) * T_ + t) * 64 + dir * 32 + n] = (_Float16)hh;
    }
  }
}

// ---------------------------------------------------------------------------
// K4: LSTM layer 1 (input K=64). Also writes h_out f32 and final hT/cT.
// ---------------------------------------------------------------------------
__global__ __launch_bounds__(128) void k_lstm1(
    const _Float16* __restrict__ x1,        // [B][T][64] f16
    const float* __restrict__ rnn1_wih,     // [2][128][64]
    const float* __restrict__ rnn_whh,      // [2][2][128][32]
    const float* __restrict__ rnn_bih,
    const float* __restrict__ rnn_bhh,
    const float* __restrict__ h0, const float* __restrict__ c0,
    float* __restrict__ hout,               // [B][T][64] f32
    float* __restrict__ hn, float* __restrict__ cn)  // [2][256][32]
{
  const int dir = blockIdx.y;
  const int b0 = blockIdx.x * 16;
  const int tid = threadIdx.x, lane = tid & 31, wid = tid >> 5;
  const float* wih = rnn1_wih + dir * 128 * 64;
  const float* whh = rnn_whh + (2 + dir) * 128 * 32;    // layer 1
  const float* bih = rnn_bih + (2 + dir) * 128;
  const float* bhh = rnn_bhh + (2 + dir) * 128;

  __shared__ _Float16 hbuf[16 * 32];
  __shared__ float    gbuf[16 * 128];

  const int n0 = wid * 32;
  const int r = lane & 15, hi = lane >> 4;
  v16h Bx00 = fragB_f32cvt(wih + n0 * 64, 64, lane);          // N-tile0, K 0..31
  v16h Bx01 = fragB_f32cvt(wih + n0 * 64 + 32, 64, lane);     // N-tile0, K 32..63
  v16h Bx10 = fragB_f32cvt(wih + (n0 + 16) * 64, 64, lane);
  v16h Bx11 = fragB_f32cvt(wih + (n0 + 16) * 64 + 32, 64, lane);
  v16h Bh0  = fragB_f32cvt(whh + n0 * 32, 32, lane);
  v16h Bh1  = fragB_f32cvt(whh + (n0 + 16) * 32, 32, lane);
  float bias0 = bih[n0 + r] + bhh[n0 + r];
  float bias1 = bih[n0 + 16 + r] + bhh[n0 + 16 + r];

  const int e0 = tid * 4, em = e0 >> 5, en = e0 & 31;
  float c[4], hl[4];
  const float* c0p = c0 + ((2 + dir) * B_ + b0) * 32;
  const float* h0p = h0 + ((2 + dir) * B_ + b0) * 32;
#pragma unroll
  for (int j = 0; j < 4; ++j) {
    c[j] = c0p[em * 32 + en + j];
    hl[j] = h0p[em * 32 + en + j];
    hbuf[em * 32 + en + j] = (_Float16)hl[j];
  }

  for (int s = 0; s < T_; ++s) {
    const int t = dir ? (T_ - 1 - s) : s;
    __syncthreads();
    v16h Ax0 = fragA_f16(x1 + ((long)b0 * T_ + t) * 64, (long)T_ * 64, lane);
    v16h Ax1 = fragA_f16(x1 + ((long)b0 * T_ + t) * 64 + 32, (long)T_ * 64, lane);
    v16h Ah  = fragA_f16(hbuf, 32, lane);
    v8f a0 = {}; v8f a1 = {};
    a0 = wmma16(Ax0, Bx00, a0);
    a1 = wmma16(Ax0, Bx10, a1);
    a0 = wmma16(Ax1, Bx01, a0);
    a1 = wmma16(Ax1, Bx11, a1);
    a0 = wmma16(Ah, Bh0, a0);
    a1 = wmma16(Ah, Bh1, a1);
#pragma unroll
    for (int rr = 0; rr < 8; ++rr) {
      gbuf[(rr + 8 * hi) * 128 + n0 + r]      = a0[rr] + bias0;
      gbuf[(rr + 8 * hi) * 128 + n0 + 16 + r] = a1[rr] + bias1;
    }
    __syncthreads();
#pragma unroll
    for (int j = 0; j < 4; ++j) {
      int n = en + j;
      float gi = gbuf[em * 128 + n];
      float gf = gbuf[em * 128 + 32 + n];
      float gg = gbuf[em * 128 + 64 + n];
      float go = gbuf[em * 128 + 96 + n];
      float cc = sigf(gf) * c[j] + sigf(gi) * tanhf(gg);
      float hh = sigf(go) * tanhf(cc);
      c[j] = cc; hl[j] = hh;
      hbuf[em * 32 + n] = (_Float16)hh;
      hout[((long)(b0 + em) * T_ + t) * 64 + dir * 32 + n] = hh;
    }
  }
#pragma unroll
  for (int j = 0; j < 4; ++j) {
    hn[(dir * B_ + b0 + em) * 32 + en + j] = hl[j];
    cn[(dir * B_ + b0 + em) * 32 + en + j] = c[j];
  }
}

// ---------------------------------------------------------------------------
// K5: attention-head input projection: prehead[k][b][t][g] = ar_wih[k][g] . hout[b][t]
// ---------------------------------------------------------------------------
__global__ __launch_bounds__(256) void k_headproj(
    const float* __restrict__ hout, const float* __restrict__ ar_wih,
    float* __restrict__ prehead) {
  __shared__ float w[16 * 64];
  for (int i = threadIdx.x; i < 1024; i += 256) w[i] = ar_wih[i];
  __syncthreads();
  long idx = (long)blockIdx.x * 256 + threadIdx.x;   // b*T + t  (131072 total)
  const float* hp = hout + idx * 64;
  float hv[64];
#pragma unroll
  for (int d = 0; d < 64; ++d) hv[d] = hp[d];
  long b = idx >> 9, t = idx & (T_ - 1);
#pragma unroll
  for (int k = 0; k < 4; ++k)
#pragma unroll
    for (int g = 0; g < 4; ++g) {
      float s = 0.0f;
#pragma unroll
      for (int d = 0; d < 64; ++d) s += w[(k * 4 + g) * 64 + d] * hv[d];
      prehead[(((long)k * B_ + b) * T_ + t) * 4 + g] = s;
    }
}

// ---------------------------------------------------------------------------
// K6: 1024 scalar LSTM recurrences (hidden=1).  heads 0,1 use l1_bwd finals,
// heads 2,3 use l1_fwd finals.
// ---------------------------------------------------------------------------
__global__ __launch_bounds__(256) void k_headrec(
    const float* __restrict__ prehead,
    const float* __restrict__ hn, const float* __restrict__ cn,
    const float* __restrict__ adh_w, const float* __restrict__ adh_b,
    const float* __restrict__ adc_w, const float* __restrict__ adc_b,
    const float* __restrict__ ar_whh, const float* __restrict__ ar_bih,
    const float* __restrict__ ar_bhh, float* __restrict__ attn) {
  int k = blockIdx.x, b = threadIdx.x;
  int sel = (k < 2) ? 1 : 0;                        // 1 = bwd finals
  const float* hsel = hn + (sel * B_ + b) * 32;
  const float* csel = cn + (sel * B_ + b) * 32;
  float h = adh_b[k], c = adc_b[k];
#pragma unroll
  for (int d = 0; d < 32; ++d) {
    h += adh_w[k * 32 + d] * hsel[d];
    c += adc_w[k * 32 + d] * csel[d];
  }
  float wi = ar_whh[k * 4 + 0], wf = ar_whh[k * 4 + 1];
  float wg = ar_whh[k * 4 + 2], wo = ar_whh[k * 4 + 3];
  float bi = ar_bih[k * 4 + 0] + ar_bhh[k * 4 + 0];
  float bf = ar_bih[k * 4 + 1] + ar_bhh[k * 4 + 1];
  float bg = ar_bih[k * 4 + 2] + ar_bhh[k * 4 + 2];
  float bo = ar_bih[k * 4 + 3] + ar_bhh[k * 4 + 3];
  const float* pp = prehead + ((long)k * B_ + b) * T_ * 4;
  float* ap = attn + ((long)k * B_ + b) * T_;
  for (int t = 0; t < T_; ++t) {
    float gi = pp[t * 4 + 0] + h * wi + bi;
    float gf = pp[t * 4 + 1] + h * wf + bf;
    float gg = pp[t * 4 + 2] + h * wg + bg;
    float go = pp[t * 4 + 3] + h * wo + bo;
    c = sigf(gf) * c + sigf(gi) * tanhf(gg);
    h = sigf(go) * tanhf(c);
    ap[t] = h;
  }
}

// ---------------------------------------------------------------------------
// K7: softmax over time + weighted pooling. one block per batch row.
// ---------------------------------------------------------------------------
__global__ __launch_bounds__(128) void k_pool(
    const float* __restrict__ attn, const float* __restrict__ hout,
    float* __restrict__ pooled) {
  __shared__ float wbuf[T_];
  __shared__ float red[128];
  int b = blockIdx.x, tid = threadIdx.x;
  for (int k = 0; k < 4; ++k) {
    const float* ap = attn + ((long)k * B_ + b) * T_;
    __builtin_prefetch(ap + T_, 0, 1);              // next batch row's logits, same head
    float m = -1e30f;
#pragma unroll
    for (int j = 0; j < 4; ++j) m = fmaxf(m, ap[tid * 4 + j]);
    red[tid] = m;
    __syncthreads();
    for (int s = 64; s > 0; s >>= 1) {
      if (tid < s) red[tid] = fmaxf(red[tid], red[tid + s]);
      __syncthreads();
    }
    float mx = red[0];
    __syncthreads();
    float ls = 0.0f;
#pragma unroll
    for (int j = 0; j < 4; ++j) {
      float e = __expf(ap[tid * 4 + j] - mx);
      wbuf[tid * 4 + j] = e;
      ls += e;
    }
    red[tid] = ls;
    __syncthreads();
    for (int s = 64; s > 0; s >>= 1) {
      if (tid < s) red[tid] += red[tid + s];
      __syncthreads();
    }
    float inv = 1.0f / red[0];
    __syncthreads();
    int n = tid & 63, half = tid >> 6;
    float acc = 0.0f;
    const float* hp = hout + ((long)b * T_ + half * 256) * 64 + n;
    for (int t = 0; t < 256; ++t) acc += wbuf[half * 256 + t] * hp[t * 64];
    red[tid] = acc;
    __syncthreads();
    if (tid < 64) pooled[((long)k * B_ + b) * 64 + n] = (red[tid] + red[tid + 64]) * inv;
    __syncthreads();
  }
}

// ---------------------------------------------------------------------------
// K8: MLP 354 -> 128 -> 128 -> 16, sigmoid out. one block per batch row.
// ---------------------------------------------------------------------------
__global__ __launch_bounds__(128) void k_mlp(
    const float* __restrict__ pooled, const float* __restrict__ xcf,
    const float* __restrict__ xcl,
    const float* __restrict__ p1w, const float* __restrict__ p1b,
    const float* __restrict__ p2w, const float* __restrict__ p2b,
    const float* __restrict__ p3w, const float* __restrict__ p3b,
    float* __restrict__ out) {
  __shared__ float inv[354];
  __shared__ float s1[128];
  __shared__ float s2[128];
  int b = blockIdx.x, tid = threadIdx.x;
  for (int d = tid; d < 354; d += 128) {
    float v;
    if (d < 256)      v = pooled[((long)(d >> 6) * B_ + b) * 64 + (d & 63)];
    else if (d < 305) v = xcf[b * 49 + (d - 256)];
    else              v = xcl[b * 49 + (d - 305)];
    inv[d] = v;
  }
  __syncthreads();
  float s = p1b[tid];
  for (int d = 0; d < 354; ++d) s += p1w[tid * 354 + d] * inv[d];
  s1[tid] = fmaxf(s, 0.0f);
  __syncthreads();
  s = p2b[tid];
#pragma unroll 4
  for (int d = 0; d < 128; ++d) s += p2w[tid * 128 + d] * s1[d];
  s2[tid] = fmaxf(s, 0.0f);
  __syncthreads();
  if (tid < 16) {
    s = p3b[tid];
#pragma unroll 4
    for (int d = 0; d < 128; ++d) s += p3w[tid * 128 + d] * s2[d];
    out[b * 16 + tid] = 1.0f / (1.0f + __expf(-s));
  }
}

// ---------------------------------------------------------------------------
extern "C" void kernel_launch(void* const* d_in, const int* in_sizes, int n_in,
                              void* d_out, int out_size, void* d_ws, size_t ws_size,
                              hipStream_t stream) {
  const float* x_tag    = (const float*)d_in[0];
  const float* xcf      = (const float*)d_in[1];
  const float* xcl      = (const float*)d_in[2];
  const float* pre_w    = (const float*)d_in[3];
  const float* pre_b    = (const float*)d_in[4];
  const float* h0_w     = (const float*)d_in[5];
  const float* h0_b     = (const float*)d_in[6];
  const float* c0_w     = (const float*)d_in[7];
  const float* c0_b     = (const float*)d_in[8];
  const float* rnn0_wih = (const float*)d_in[9];
  const float* rnn1_wih = (const float*)d_in[10];
  const float* rnn_whh  = (const float*)d_in[11];
  const float* rnn_bih  = (const float*)d_in[12];
  const float* rnn_bhh  = (const float*)d_in[13];
  const float* adh_w    = (const float*)d_in[14];
  const float* adh_b    = (const float*)d_in[15];
  const float* adc_w    = (const float*)d_in[16];
  const float* adc_b    = (const float*)d_in[17];
  const float* ar_wih   = (const float*)d_in[18];
  const float* ar_whh   = (const float*)d_in[19];
  const float* ar_bih   = (const float*)d_in[20];
  const float* ar_bhh   = (const float*)d_in[21];
  const float* p1_w     = (const float*)d_in[22];
  const float* p1_b     = (const float*)d_in[23];
  const float* p2_w     = (const float*)d_in[24];
  const float* p2_b     = (const float*)d_in[25];
  const float* p3_w     = (const float*)d_in[26];
  const float* p3_b     = (const float*)d_in[27];
  float* out = (float*)d_out;

  char* ws = (char*)d_ws;
  float*     h0o     = (float*)(ws + 0);              //  131072 B
  float*     c0o     = (float*)(ws + 131072);         //  131072 B
  _Float16*  htag    = (_Float16*)(ws + 262144);      //  8 MB
  _Float16*  x1      = (_Float16*)(ws + 8650752);     // 16 MB
  float*     hout    = (float*)(ws + 25427968);       // 32 MB
  float*     hn      = (float*)(ws + 58982400);       // 64 KB
  float*     cn      = (float*)(ws + 59047936);       // 64 KB
  float*     prehead = (float*)(ws + 59113472);       //  8 MB
  float*     attn    = (float*)(ws + 67502080);       //  2 MB
  float*     pooled  = (float*)(ws + 69599232);       // 256 KB  (total ~66.7 MB)

  k_init<<<128, 256, 0, stream>>>(xcf, xcl, h0_w, h0_b, c0_w, c0_b, h0o, c0o);
  k_htag<<<1024, 256, 0, stream>>>(x_tag, pre_w, pre_b, htag);
  k_lstm0<<<dim3(16, 2), 128, 0, stream>>>(htag, rnn0_wih, rnn_whh, rnn_bih, rnn_bhh,
                                           h0o, c0o, x1);
  k_lstm1<<<dim3(16, 2), 128, 0, stream>>>(x1, rnn1_wih, rnn_whh, rnn_bih, rnn_bhh,
                                           h0o, c0o, hout, hn, cn);
  k_headproj<<<512, 256, 0, stream>>>(hout, ar_wih, prehead);
  k_headrec<<<4, 256, 0, stream>>>(prehead, hn, cn, adh_w, adh_b, adc_w, adc_b,
                                   ar_whh, ar_bih, ar_bhh, attn);
  k_pool<<<256, 128, 0, stream>>>(attn, hout, pooled);
  k_mlp<<<256, 128, 0, stream>>>(pooled, xcf, xcl, p1_w, p1_b, p2_w, p2_b, p3_w, p3_b, out);
}
